// MoCWrapper_16947940950232
// MI455X (gfx1250) — compile-verified, matrix operand order checked by reading, not run on hardware
//
#include <hip/hip_runtime.h>

// Model dims
#define NL   16
#define NT   2048
#define ND   2048
#define NR   128
#define KCAP 1024

typedef __attribute__((ext_vector_type(16))) __bf16 v16bf;
typedef __attribute__((ext_vector_type(8)))  __bf16 v8bf;
typedef __attribute__((ext_vector_type(4)))  __bf16 v4bf;
typedef __attribute__((ext_vector_type(8)))  float  v8f;
typedef __attribute__((ext_vector_type(4)))  float  v4f;   // trivial type: addrspace-cast OK

// GEMM tiling: 128x128 output tile per 256-thread (8-wave) workgroup, K-step 32.
#define BM   128
#define BN   128
#define BK   32
#define ASTR 40   // padded LDS row stride (bf16 elems): 80B rows -> 16B aligned, conflict-light

#if defined(__has_builtin) && __has_builtin(__builtin_amdgcn_sched_barrier)
#define SCHED_FENCE() __builtin_amdgcn_sched_barrier(0)
#else
#define SCHED_FENCE()
#endif

// GLOBAL (addrspace 1) load: uniform SGPR base + 32-bit zext VGPR offset -> saddr form.
__device__ __forceinline__ v4f gload4(const float* __restrict__ base, unsigned off) {
  return *(const __attribute__((address_space(1))) v4f*)(base + off);
}

// WMMA A/B fragment (16x32 bf16) from LDS, per CDNA5 ISA layout:
// lane: m = lane&15, h = lane>>4; elems 0..7 = [m, 8h+0..7], elems 8..15 = [m, 16+8h+0..7]
__device__ __forceinline__ v16bf frag_ld(const __bf16* base, int lane) {
  const int m = lane & 15;
  const int h = lane >> 4;
  union { v16bf v; v8bf q[2]; } u;
  u.q[0] = *reinterpret_cast<const v8bf*>(base + m * ASTR + 8 * h);        // ds_load_b128
  u.q[1] = *reinterpret_cast<const v8bf*>(base + m * ASTR + 16 + 8 * h);   // ds_load_b128
  return u.v;
}

// Convert fp32 -> bf16 (native v_cvt_pk_bf16_f32) and store into LDS;
// A row-major [row][k], B transposed [n][k]
__device__ __forceinline__ void stage_store(__bf16* Asb, __bf16* Bsb,
                                            const v4f (&ra)[4], const v4f (&rb)[4],
                                            int tid) {
#pragma unroll
  for (int i = 0; i < 4; ++i) {
    const int idx = tid + i * 256;
    {
      const int row = idx >> 3, c4 = idx & 7;
      v4bf p = { (__bf16)ra[i].x, (__bf16)ra[i].y, (__bf16)ra[i].z, (__bf16)ra[i].w };
      *reinterpret_cast<v4bf*>(Asb + row * ASTR + c4 * 4) = p;             // ds_store_b64
    }
    {
      const int k  = idx >> 5;
      const int n4 = (idx & 31) * 4;
      Bsb[(n4 + 0) * ASTR + k] = (__bf16)rb[i].x;
      Bsb[(n4 + 1) * ASTR + k] = (__bf16)rb[i].y;
      Bsb[(n4 + 2) * ASTR + k] = (__bf16)rb[i].z;
      Bsb[(n4 + 3) * ASTR + k] = (__bf16)rb[i].w;
    }
  }
}

// Double-buffered bf16 WMMA GEMM core: acc[2][4] covers a 32x64 sub-tile per wave.
__device__ __forceinline__ void gemm_bf16(const float* __restrict__ Abase,
                                          const float* __restrict__ Bbase, int ldb,
                                          __bf16 (*As)[BM * ASTR],
                                          __bf16 (*Bs)[BN * ASTR],
                                          v8f (&acc)[2][4],
                                          int tid, int lane, int wr, int wc) {
  // Per-thread constant 32-bit element offsets (max ND*NT = 4M < 2^31).
  unsigned aoff[4], boff[4];
#pragma unroll
  for (int i = 0; i < 4; ++i) {
    const int idx = tid + i * 256;  // 0..1023 float4 groups
    aoff[i] = (unsigned)((idx >> 3) * ND + (idx & 7) * 4);
    boff[i] = (unsigned)((idx >> 5) * ldb + (idx & 31) * 4);
  }
  v4f ra[4], rb[4];
#pragma unroll
  for (int i = 0; i < 4; ++i) {
    ra[i] = gload4(Abase, aoff[i]);
    rb[i] = gload4(Bbase, boff[i]);
  }
  stage_store(As[0], Bs[0], ra, rb, tid);
  __syncthreads();

  const int NKS = ND / BK;  // 64
  for (int ks = 0; ks < NKS; ++ks) {
    const int buf = ks & 1;
    if (ks + 1 < NKS) {
      const unsigned ka = (unsigned)((ks + 1) * BK);        // uniform (SALU) K advance
      const unsigned kb = (unsigned)((ks + 1) * BK * ldb);
#pragma unroll
      for (int i = 0; i < 4; ++i) {
        ra[i] = gload4(Abase, aoff[i] + ka);
        rb[i] = gload4(Bbase, boff[i] + kb);
      }
    }
    // Pin the prefetch loads BEFORE the WMMA block so HBM/L2 latency overlaps compute;
    // without this the scheduler sinks them after the WMMAs and serializes load->cvt.
    SCHED_FENCE();
    const __bf16* Ab = As[buf];
    const __bf16* Bb = Bs[buf];
    v16bf a0 = frag_ld(Ab + (wr * 32) * ASTR, lane);
    v16bf a1 = frag_ld(Ab + (wr * 32 + 16) * ASTR, lane);
    v16bf bf[4];
#pragma unroll
    for (int ct = 0; ct < 4; ++ct)
      bf[ct] = frag_ld(Bb + (wc * 64 + ct * 16) * ASTR, lane);
#pragma unroll
    for (int ct = 0; ct < 4; ++ct) {
      acc[0][ct] = __builtin_amdgcn_wmma_f32_16x16x32_bf16(false, a0, false, bf[ct],
                                                           (short)0, acc[0][ct], false, false);
      acc[1][ct] = __builtin_amdgcn_wmma_f32_16x16x32_bf16(false, a1, false, bf[ct],
                                                           (short)0, acc[1][ct], false, false);
    }
    if (ks + 1 < NKS)
      stage_store(As[buf ^ 1], Bs[buf ^ 1], ra, rb, tid);
    __syncthreads();
  }
}

// ---------------- Kernel 1: router probs = sigmoid(relu(hidden@Wr1 + br1)@Wr2 + br2) ----
__global__ __launch_bounds__(256) void moc_router(
    const float* __restrict__ hidden, const float* __restrict__ Wr1,
    const float* __restrict__ br1, const float* __restrict__ Wr2,
    const float* __restrict__ br2, float* __restrict__ probs) {
  __shared__ __align__(16) __bf16 As[2][BM * ASTR];
  __shared__ __align__(16) __bf16 Bs[2][BN * ASTR];
  __shared__ float ps[BM][2];

  const int l   = blockIdx.x >> 4;
  const int tm  = blockIdx.x & 15;
  const int tid = threadIdx.x, lane = tid & 31, wave = tid >> 5;
  const int wr  = wave >> 1, wc = wave & 1;

  const float* Abase = hidden + ((size_t)l * NT + (size_t)tm * BM) * ND;
  const float* Bbase = Wr1 + (size_t)l * ND * NR;

  v8f acc[2][4] = {};
  gemm_bf16(Abase, Bbase, NR, As, Bs, acc, tid, lane, wr, wc);

  float biasv[4], wv[4];
#pragma unroll
  for (int ct = 0; ct < 4; ++ct) {
    const int col = wc * 64 + ct * 16 + (lane & 15);
    biasv[ct] = br1[l * NR + col];
    wv[ct]    = Wr2[l * NR + col];
  }
#pragma unroll
  for (int rt = 0; rt < 2; ++rt) {
#pragma unroll
    for (int j = 0; j < 8; ++j) {
      float s = 0.0f;
#pragma unroll
      for (int ct = 0; ct < 4; ++ct) {
        float h = acc[rt][ct][j] + biasv[ct];
        h = h > 0.0f ? h : 0.0f;       // relu
        s += h * wv[ct];
      }
#pragma unroll
      for (int off = 1; off < 16; off <<= 1) s += __shfl_xor(s, off, 32);
      const int row = wr * 32 + rt * 16 + j + 8 * (lane >> 4);
      if ((lane & 15) == 0) ps[row][wc] = s;
    }
  }
  __syncthreads();
  if (tid < BM) {
    const float logit = ps[tid][0] + ps[tid][1] + br2[l];
    probs[(size_t)l * NT + (size_t)tm * BM + tid] = 1.0f / (1.0f + __expf(-logit));
  }
}

// ---------------- Kernel 2: per-layer top-K threshold + straight-through mask ----------
__global__ __launch_bounds__(1024) void moc_topk(const float* __restrict__ probs,
                                                 float* __restrict__ mst) {
  __shared__ float v[NT];
  const int l   = blockIdx.x;
  const int tid = threadIdx.x;
  const float* p = probs + (size_t)l * NT;
  v[tid] = p[tid];
  v[tid + 1024] = p[tid + 1024];
  __syncthreads();
  // bitonic sort ascending (2048 elements, 1024 threads, 1 pair/thread/step)
  for (int k = 2; k <= NT; k <<= 1) {
    for (int j = k >> 1; j > 0; j >>= 1) {
      const int i   = ((tid & ~(j - 1)) << 1) | (tid & (j - 1));
      const int ixj = i | j;
      const float a = v[i], b = v[ixj];
      const bool up = ((i & k) == 0);
      if ((a > b) == up) { v[i] = b; v[ixj] = a; }
      __syncthreads();
    }
  }
  const float thr = v[NT - KCAP];  // K-th largest
  for (int t = tid; t < NT; t += 1024) {
    const float pv = p[t];
    // mask - sg(p) + p evaluated left-to-right, matching the reference's fp order
    const float m = (pv >= thr) ? ((1.0f - pv) + pv) : 0.0f;
    mst[(size_t)l * NT + t] = m;
  }
}

// ---------------- Kernel 3: s1 = hidden@Ws1 + bs1, fused blend with s2 -----------------
__global__ __launch_bounds__(256) void moc_s1_blend(
    const float* __restrict__ hidden, const float* __restrict__ s2,
    const float* __restrict__ Ws1, const float* __restrict__ bs1,
    const float* __restrict__ mst, float* __restrict__ out) {
  __shared__ __align__(16) __bf16 As[2][BM * ASTR];
  __shared__ __align__(16) __bf16 Bs[2][BN * ASTR];

  const int bid = blockIdx.x;
  const int l   = bid >> 8;
  const int tm  = (bid >> 4) & 15;
  const int tn  = bid & 15;
  const int tid = threadIdx.x, lane = tid & 31, wave = tid >> 5;
  const int wr  = wave >> 1, wc = wave & 1;

  const float* Abase = hidden + ((size_t)l * NT + (size_t)tm * BM) * ND;
  const float* Bbase = Ws1 + (size_t)l * ND * ND + (size_t)tn * BN;

  v8f acc[2][4] = {};
  gemm_bf16(Abase, Bbase, ND, As, Bs, acc, tid, lane, wr, wc);

  const int row0 = tm * BM + wr * 32;
  const int col0 = tn * BN + wc * 64;
  float bv[4];
#pragma unroll
  for (int ct = 0; ct < 4; ++ct)
    bv[ct] = bs1[(size_t)l * ND + col0 + ct * 16 + (lane & 15)];

#pragma unroll
  for (int rt = 0; rt < 2; ++rt) {
#pragma unroll
    for (int j = 0; j < 8; ++j) {
      const int row  = row0 + rt * 16 + j + 8 * (lane >> 4);
      const float m  = mst[(size_t)l * NT + row];     // one load per row
      const size_t rbase = ((size_t)l * NT + row) * ND;
#pragma unroll
      for (int ct = 0; ct < 4; ++ct) {
        const int col  = col0 + ct * 16 + (lane & 15);
        const size_t o = rbase + col;
        const float s1 = acc[rt][ct][j] + bv[ct];
        out[o] = m * s2[o] + (1.0f - m) * s1;
      }
    }
  }
}

extern "C" void kernel_launch(void* const* d_in, const int* in_sizes, int n_in,
                              void* d_out, int out_size, void* d_ws, size_t ws_size,
                              hipStream_t stream) {
  (void)in_sizes; (void)n_in; (void)out_size; (void)ws_size;
  const float* hidden = (const float*)d_in[0];
  const float* s2     = (const float*)d_in[1];
  const float* Wr1    = (const float*)d_in[2];
  const float* br1    = (const float*)d_in[3];
  const float* Wr2    = (const float*)d_in[4];
  const float* br2    = (const float*)d_in[5];
  const float* Ws1    = (const float*)d_in[6];
  const float* bs1    = (const float*)d_in[7];
  float* out   = (float*)d_out;
  float* probs = (float*)d_ws;               // NL*NT floats
  float* mst   = probs + (size_t)NL * NT;    // NL*NT floats  (256 KB total workspace)

  moc_router<<<NL * (NT / BM), 256, 0, stream>>>(hidden, Wr1, br1, Wr2, br2, probs);
  moc_topk<<<NL, 1024, 0, stream>>>(probs, mst);
  moc_s1_blend<<<NL * (NT / BM) * (ND / BN), 256, 0, stream>>>(hidden, s2, Ws1, bs1, mst, out);
}
